// FastKMeansClassifier_32246614458630
// MI455X (gfx1250) — compile-verified, matrix operand order checked by reading, not run on hardware
//
#include <hip/hip_runtime.h>

// ---------------------------------------------------------------------------
// FastKMeansClassifier fused kernel for MI455X (gfx1250, wave32, WMMA)
//   out[n, cls] = softmax_over_centroids(X @ C^T)[n, :] bucketed by labels
// Single GEMM pass; block-uniform online bound (cancels in per-row softmax
// normalization) with rare uniform rescale of LDS class buckets.
// ---------------------------------------------------------------------------

typedef __bf16 v8bf  __attribute__((ext_vector_type(8)));
typedef __bf16 v16bf __attribute__((ext_vector_type(16)));
typedef float  v8f   __attribute__((ext_vector_type(8)));

#define DIM      1024
#define ROWS     16
#define CLASSES  4096
#define NCENT    12288
#define CHUNK    256          // centroids per chunk = 8 waves * 2 tiles * 16 cols
#define NCHUNK   (NCENT / CHUNK)
#define XPAD     (DIM + 8)    // 2064 B row stride: 16B aligned, banks staggered

// ---- pass 0: convert centroids f32 -> bf16 into workspace (24 MB, L2-resident)
__global__ __launch_bounds__(256)
void convert_to_bf16(const float* __restrict__ s, __bf16* __restrict__ d, int n8) {
    int i = blockIdx.x * blockDim.x + threadIdx.x;
    if (i >= n8) return;
    const float4* s4 = (const float4*)s;
    float4 f0 = s4[2 * i + 0];
    float4 f1 = s4[2 * i + 1];
    v8bf o;
    o[0] = (__bf16)f0.x; o[1] = (__bf16)f0.y; o[2] = (__bf16)f0.z; o[3] = (__bf16)f0.w;
    o[4] = (__bf16)f1.x; o[5] = (__bf16)f1.y; o[6] = (__bf16)f1.z; o[7] = (__bf16)f1.w;
    ((v8bf*)d)[i] = o;
}

// ---- pass 1: fused GEMM + online softmax + label bucketing + normalize
__global__ __launch_bounds__(256)
void fused_kmeans_softmax(const float* __restrict__ X,
                          const __bf16* __restrict__ Cb,
                          const int* __restrict__ labels,
                          float* __restrict__ out)
{
    __shared__ float  s_acc[ROWS][CLASSES];   // 256 KB class buckets
    __shared__ __bf16 s_x[ROWS][XPAD];        // 32.25 KB X tile (bf16)
    __shared__ float  s_wavemax[8];
    __shared__ float  s_bound;                // running uniform upper bound
    __shared__ float  s_scale;
    __shared__ int    s_need;
    __shared__ float  s_rinv[ROWS];

    const int tid  = threadIdx.x;
    const int wave = tid >> 5;
    const int lane = tid & 31;
    const int half = lane >> 4;    // 0: lanes 0-15, 1: lanes 16-31
    const int sub  = lane & 15;    // A-row (M) and B-column (N) index
    const long rowbase = (long)blockIdx.x * ROWS;

    // Stage X rows into LDS as bf16
    for (int idx = tid; idx < ROWS * DIM; idx += 256) {
        int r = idx >> 10, k = idx & (DIM - 1);
        s_x[r][k] = (__bf16)X[(rowbase + r) * DIM + k];
    }
    // Zero class buckets
    for (int idx = tid; idx < ROWS * CLASSES; idx += 256)
        ((float*)s_acc)[idx] = 0.0f;
    if (tid == 0) s_bound = -3.0e38f;
    __syncthreads();

    const __bf16* xrow = &s_x[sub][0];
    const int k0 = half * 8;   // 16-bit operand layout: lanes 0-15 -> K 0-7/16-23,
                               //                        lanes 16-31 -> K 8-15/24-31

    for (int chunk = 0; chunk < NCHUNK; ++chunk) {
        const int cent0 = chunk * CHUNK + wave * 16 + sub;         // tile 0 column
        const int cent1 = cent0 + 128;                             // tile 1 column
        const __bf16* crow0 = Cb + (long)cent0 * DIM;
        const __bf16* crow1 = Cb + (long)cent1 * DIM;

        if (chunk + 1 < NCHUNK) {   // speculative warm-up of next chunk's rows
            __builtin_prefetch(crow0 + (long)CHUNK * DIM, 0, 0);
            __builtin_prefetch(crow1 + (long)CHUNK * DIM, 0, 0);
        }

        // ---- two independent 16x16 logit tiles, K = 1024 (2x32 chained WMMAs)
        v8f acc0 = {}, acc1 = {};
        #pragma unroll 4
        for (int kk = 0; kk < DIM; kk += 32) {
            v8bf alo = *(const v8bf*)(xrow + kk + k0);
            v8bf ahi = *(const v8bf*)(xrow + kk + k0 + 16);
            v16bf a = __builtin_shufflevector(alo, ahi, 0,1,2,3,4,5,6,7,8,9,10,11,12,13,14,15);
            v8bf b0lo = *(const v8bf*)(crow0 + kk + k0);
            v8bf b0hi = *(const v8bf*)(crow0 + kk + k0 + 16);
            v8bf b1lo = *(const v8bf*)(crow1 + kk + k0);
            v8bf b1hi = *(const v8bf*)(crow1 + kk + k0 + 16);
            v16bf b0 = __builtin_shufflevector(b0lo, b0hi, 0,1,2,3,4,5,6,7,8,9,10,11,12,13,14,15);
            v16bf b1 = __builtin_shufflevector(b1lo, b1hi, 0,1,2,3,4,5,6,7,8,9,10,11,12,13,14,15);
            acc0 = __builtin_amdgcn_wmma_f32_16x16x32_bf16(
                       false, a, false, b0, (short)0, acc0, false, false);
            acc1 = __builtin_amdgcn_wmma_f32_16x16x32_bf16(
                       false, a, false, b1, (short)0, acc1, false, false);
        }

        // ---- wave-wide chunk bound (VALU max tree + 5 lane shuffles)
        float bmax = acc0[0];
        #pragma unroll
        for (int i = 1; i < 8; ++i) bmax = fmaxf(bmax, acc0[i]);
        #pragma unroll
        for (int i = 0; i < 8; ++i) bmax = fmaxf(bmax, acc1[i]);
        bmax = fmaxf(bmax, __shfl_xor(bmax, 1));
        bmax = fmaxf(bmax, __shfl_xor(bmax, 2));
        bmax = fmaxf(bmax, __shfl_xor(bmax, 4));
        bmax = fmaxf(bmax, __shfl_xor(bmax, 8));
        bmax = fmaxf(bmax, __shfl_xor(bmax, 16));
        if (lane == 0) s_wavemax[wave] = bmax;
        __syncthreads();

        // ---- update uniform bound; scalar rescale factor
        if (tid == 0) {
            float cb = s_wavemax[0];
            #pragma unroll
            for (int w = 1; w < 8; ++w) cb = fmaxf(cb, s_wavemax[w]);
            float old = s_bound;
            float nm  = fmaxf(old, cb);
            s_scale = __expf(old - nm);    // 0 on first chunk, else <= 1
            s_need  = (nm > old) ? 1 : 0;
            s_bound = nm;
        }
        __syncthreads();

        if (s_need) {                      // rare (~ln(NCHUNK) times total)
            const float sc = s_scale;
            for (int idx = tid; idx < ROWS * CLASSES; idx += 256)
                ((float*)s_acc)[idx] *= sc;
            __syncthreads();
        }

        // ---- exp and scatter into class buckets (ds_add_f32)
        const float m   = s_bound;
        const int   lab0 = labels[cent0];
        const int   lab1 = labels[cent1];
        #pragma unroll
        for (int i = 0; i < 8; ++i) {
            int row = half * 8 + i;
            atomicAdd(&s_acc[row][lab0], __expf(acc0[i] - m));
            atomicAdd(&s_acc[row][lab1], __expf(acc1[i] - m));
        }
    }
    __syncthreads();

    // ---- softmax denominator = sum of all buckets in a row (e^-m cancels)
    {
        int r = tid >> 4;
        int j = tid & 15;
        float s = 0.0f;
        for (int c = j; c < CLASSES; c += 16) s += s_acc[r][c];
        s += __shfl_xor(s, 1);
        s += __shfl_xor(s, 2);
        s += __shfl_xor(s, 4);
        s += __shfl_xor(s, 8);
        if (j == 0) s_rinv[r] = 1.0f / s;
    }
    __syncthreads();

    // ---- normalized f32 output
    for (int idx = tid; idx < ROWS * CLASSES; idx += 256) {
        int r = idx >> 12, c = idx & (CLASSES - 1);
        out[(rowbase + r) * CLASSES + c] = s_acc[r][c] * s_rinv[r];
    }
}

extern "C" void kernel_launch(void* const* d_in, const int* in_sizes, int n_in,
                              void* d_out, int out_size, void* d_ws, size_t ws_size,
                              hipStream_t stream) {
    const float* X      = (const float*)d_in[0];   // [16384, 1024] f32
    const float* C      = (const float*)d_in[1];   // [12288, 1024] f32
    const int*   labels = (const int*)d_in[2];     // [12288] i32
    float*       out    = (float*)d_out;           // [16384, 4096] f32
    __bf16*      Cb     = (__bf16*)d_ws;           // 24 MB bf16 centroids

    (void)in_sizes; (void)n_in; (void)out_size; (void)ws_size;

    int n8 = NCENT * DIM / 8;
    convert_to_bf16<<<(n8 + 255) / 256, 256, 0, stream>>>(C, Cb, n8);

    fused_kmeans_softmax<<<16384 / ROWS, 256, 0, stream>>>(X, Cb, labels, out);
}
